// PointnetFPModule_22067541967444
// MI455X (gfx1250) — matrix operand order repaired and checked by reading, not compiled, not run on hardware
//
#include <hip/hip_runtime.h>
#include <hip/hip_fp16.h>

typedef __attribute__((ext_vector_type(16))) _Float16 v16h;
typedef __attribute__((ext_vector_type(8)))  float    v8f;

namespace {
constexpr int kB    = 4;
constexpr int kNS   = 4096;
constexpr int kNL   = 16384;
constexpr int kC1   = 256;
constexpr int kC2   = 128;
constexpr int kCin  = 384;    // C1 + C2
constexpr int kCmid = 256;    // MLP widths
constexpr int kN    = 65536;  // B * NL, flattened point count
}

// ---------------------------------------------------------------------------
// Fold Conv bias + eval-mode BatchNorm into per-channel scale/bias:
//   y = (Wx + b) * inv + (beta - rm * inv),  inv = g / sqrt(rv + eps)
//     = (Wx) * scale + bias,  scale = inv,  bias = b*inv + beta - rm*inv
// ---------------------------------------------------------------------------
__global__ void fold_bn_kernel(const float* __restrict__ b,
                               const float* __restrict__ g,
                               const float* __restrict__ be,
                               const float* __restrict__ rm,
                               const float* __restrict__ rv,
                               float* __restrict__ scale,
                               float* __restrict__ bias, int C)
{
    int c = blockIdx.x * blockDim.x + threadIdx.x;
    if (c < C) {
        float inv = g[c] * rsqrtf(rv[c] + 1e-5f);
        scale[c] = inv;
        bias[c]  = b[c] * inv + be[c] - rm[c] * inv;
    }
}

// ---------------------------------------------------------------------------
// Pack fp32 weights [O, C] into the wave32 A-fragment layout of
// v_wmma_f32_16x16x32_f16 (ISA 7.12.2, 16-bit A 16x32):
//   lane 0..15  : row M = lane,    halves 0..7 = K 0..7,  halves 8..15 = K 16..23
//   lane 16..31 : row M = lane-16, halves 0..7 = K 8..15, halves 8..15 = K 24..31
// Stored fragment-contiguous: [Mtile][Ktile][lane][16 halves] so the GEMM
// does one coalesced 32B load per lane per fragment.
// ---------------------------------------------------------------------------
__global__ void pack_w_kernel(const float* __restrict__ W,
                              _Float16* __restrict__ Wp, int O, int C)
{
    int ktiles = C >> 5;
    int total  = (O >> 4) * ktiles * 32;
    int t = blockIdx.x * blockDim.x + threadIdx.x;
    if (t >= total) return;
    int lane = t & 31;
    int tile = t >> 5;
    int kt = tile % ktiles;
    int mt = tile / ktiles;
    int m  = (mt << 4) + (lane & 15);
    _Float16* dst = Wp + (size_t)t * 16;
#pragma unroll
    for (int h = 0; h < 16; ++h) {
        int k = (lane < 16) ? ((h < 8) ? h : h + 8)
                            : ((h < 8) ? h + 8 : h + 16);
        k += (kt << 5);
        dst[h] = (_Float16)W[(size_t)m * C + k];
    }
}

// ---------------------------------------------------------------------------
// 3-NN + inverse-distance weights + max-interpolate + concat(feat_up).
// pc_down for the batch (48 KB) lives in LDS; every lane streams all 4096
// candidates (same address per iteration -> LDS broadcast, no bank conflicts).
// Writes x as f16 row-major [kCin][kN]: exactly the B-matrix source layout
// the WMMA GEMM wants (each lane gathers 16 k-strided halves of one n column).
// ---------------------------------------------------------------------------
__global__ __launch_bounds__(256) void knn_interp_kernel(
    const float* __restrict__ pc_down,   // [B,3,NS]
    const float* __restrict__ pc_up,     // [B,3,NL]
    const float* __restrict__ feat_down, // [B,C1,NS]
    const float* __restrict__ feat_up,   // [B,C2,NL]
    _Float16* __restrict__ xh)           // [kCin, kN] f16
{
    __shared__ float sx[kNS];
    __shared__ float sy[kNS];
    __shared__ float sz[kNS];

    const int b = blockIdx.y;
    const int n = blockIdx.x * 256 + threadIdx.x;

    const float* pd = pc_down + (size_t)b * 3 * kNS;
    for (int s = threadIdx.x; s < kNS; s += 256) {
        sx[s] = pd[s];
        sy[s] = pd[kNS + s];
        sz[s] = pd[2 * kNS + s];
    }
    __syncthreads();

    const float* pu = pc_up + (size_t)b * 3 * kNL;
    const float qx = pu[n];
    const float qy = pu[kNL + n];
    const float qz = pu[2 * kNL + n];

    float d0 = 3.4e38f, d1 = 3.4e38f, d2 = 3.4e38f;
    int   i0 = 0,       i1 = 0,       i2 = 0;
    for (int s = 0; s < kNS; ++s) {
        float dx = qx - sx[s];
        float dy = qy - sy[s];
        float dz = qz - sz[s];
        float d  = dx * dx + dy * dy + dz * dz;
        if (d < d0)      { d2 = d1; i2 = i1; d1 = d0; i1 = i0; d0 = d; i0 = s; }
        else if (d < d1) { d2 = d1; i2 = i1; d1 = d;  i1 = s; }
        else if (d < d2) { d2 = d;  i2 = s; }
    }

    float r0 = 1.f / (d0 + 1e-8f);
    float r1 = 1.f / (d1 + 1e-8f);
    float r2 = 1.f / (d2 + 1e-8f);
    float rs = 1.f / (r0 + r1 + r2);
    float w0 = r0 * rs, w1 = r1 * rs, w2 = r2 * rs;

    const float* fd  = feat_down + (size_t)b * kC1 * kNS;
    const size_t col = (size_t)b * kNL + n;
    for (int c = 0; c < kC1; ++c) {
        const float* fr = fd + (size_t)c * kNS;   // L2-resident gathers
        float v = fmaxf(fmaxf(w0 * fr[i0], w1 * fr[i1]), w2 * fr[i2]);
        xh[(size_t)c * kN + col] = (_Float16)v;
    }
    const float* fu = feat_up + (size_t)b * kC2 * kNL;
    for (int c = 0; c < kC2; ++c) {
        xh[(size_t)(kC1 + c) * kN + col] = (_Float16)fu[(size_t)c * kNL + n];
    }
}

// ---------------------------------------------------------------------------
// WMMA GEMM + fused BN + ReLU.  y[o,n] = relu(scale[o]*sum_k W[o,k]*x[k,n]+bias[o])
// 256 threads = 8 waves; wave tile 64(M) x 32(N) -> 4x2 v8f accumulators.
// B fragment (16-bit B 32x16, wave32): lane L holds column N = L & 15, 16 K values
// at K = 8*(L>>4)+s (s<8), 16+8*(L>>4)+s-8 (s>=8) -> 16 strided loads from [K][kN].
// C/D layout: VGPR r holds M = r + 8*(lane>=16), N = lane & 15.
//
// K and MODE are compile-time so strides constant-fold; the K-loop is
// explicitly double-buffered so WMMAs for tile kt overlap the in-flight
// loads for tile kt+1 (partial s_wait_loadcnt instead of full drain).
// MODE 0: write f16 activations (input of next GEMM).
// MODE 1: write f32 to out[B,256,NL] (col = b*NL + n; NL % 16 == 0).
// ---------------------------------------------------------------------------
template <int K, int MODE>
__global__ __launch_bounds__(256) void gemm_bn_relu_kernel(
    const _Float16* __restrict__ Ap,    // packed weights
    const _Float16* __restrict__ Bsrc,  // f16 activations [K][kN]
    const float* __restrict__ scale,
    const float* __restrict__ bias,
    _Float16* __restrict__ outH,
    float* __restrict__ outF)
{
    constexpr int ktiles = K >> 5;
    static_assert((ktiles & 1) == 0, "pipeline assumes even ktile count");

    const int lane   = threadIdx.x & 31;
    const int wave   = threadIdx.x >> 5;
    const int m_base = (wave & 3) << 6;                         // 0,64,128,192
    const int n_base = (blockIdx.x << 6) + ((wave >> 2) << 5);  // 64-wide block

    v8f acc[4][2] = {};

    const _Float16* apBase = Ap + ((size_t)(m_base >> 4) * ktiles + 0) * 32 * 16
                                + (size_t)lane * 16;
    const _Float16* bBase  = Bsrc + (size_t)((lane >> 4) << 3) * kN + n_base + (lane & 15);

    auto loadFrag = [&](int kt, v16h a[4], v16h b[2]) {
#pragma unroll
        for (int i = 0; i < 4; ++i) {
            a[i] = *(const v16h*)(apBase + ((size_t)i * ktiles + kt) * 32 * 16);
        }
        const _Float16* brow = bBase + (size_t)(kt << 5) * kN;
#pragma unroll
        for (int s = 0; s < 16; ++s) {
            const int k = (s < 8) ? s : s + 8;
            b[0][s] = brow[(size_t)k * kN];
            b[1][s] = brow[(size_t)k * kN + 16];
        }
    };
    auto domma = [&](const v16h a[4], const v16h b[2]) {
#pragma unroll
        for (int i = 0; i < 4; ++i) {
            acc[i][0] = __builtin_amdgcn_wmma_f32_16x16x32_f16(
                false, a[i], false, b[0], (short)0, acc[i][0], false, false);
            acc[i][1] = __builtin_amdgcn_wmma_f32_16x16x32_f16(
                false, a[i], false, b[1], (short)0, acc[i][1], false, false);
        }
    };

    v16h a0[4], a1[4], b0[2], b1[2];
    loadFrag(0, a0, b0);                       // prologue
#pragma unroll 1
    for (int kt = 0; kt + 2 < ktiles; kt += 2) {
        loadFrag(kt + 1, a1, b1);              // prefetch while mma(kt) runs
        domma(a0, b0);
        loadFrag(kt + 2, a0, b0);              // prefetch while mma(kt+1) runs
        domma(a1, b1);
    }
    loadFrag(ktiles - 1, a1, b1);              // epilogue of pipeline
    domma(a0, b0);
    domma(a1, b1);

    const int lane_n = lane & 15;
    const int m_hi   = (lane >> 4) << 3;  // 0 or 8
#pragma unroll
    for (int i = 0; i < 4; ++i) {
#pragma unroll
        for (int r = 0; r < 8; ++r) {
            const int   row = m_base + (i << 4) + m_hi + r;
            const float sc  = scale[row];
            const float bi  = bias[row];
#pragma unroll
            for (int j = 0; j < 2; ++j) {
                const int col = n_base + (j << 4) + lane_n;
                float v = fmaxf(acc[i][j][r] * sc + bi, 0.f);
                if (MODE == 0) {
                    outH[(size_t)row * kN + col] = (_Float16)v;
                } else {
                    int bb = col >> 14;          // col / NL
                    int nn = col & (kNL - 1);    // col % NL
                    outF[((size_t)bb * kCmid + row) * kNL + nn] = v;
                }
            }
        }
    }
}

// ---------------------------------------------------------------------------
extern "C" void kernel_launch(void* const* d_in, const int* in_sizes, int n_in,
                              void* d_out, int out_size, void* d_ws, size_t ws_size,
                              hipStream_t stream)
{
    (void)in_sizes; (void)n_in; (void)out_size; (void)ws_size;

    const float* pc_down   = (const float*)d_in[0];
    const float* pc_up     = (const float*)d_in[1];
    const float* feat_down = (const float*)d_in[2];
    const float* feat_up   = (const float*)d_in[3];
    const float* W1  = (const float*)d_in[4];
    const float* b1  = (const float*)d_in[5];
    const float* g1  = (const float*)d_in[6];
    const float* be1 = (const float*)d_in[7];
    const float* rm1 = (const float*)d_in[8];
    const float* rv1 = (const float*)d_in[9];
    const float* W2  = (const float*)d_in[10];
    const float* b2  = (const float*)d_in[11];
    const float* g2  = (const float*)d_in[12];
    const float* be2 = (const float*)d_in[13];
    const float* rm2 = (const float*)d_in[14];
    const float* rv2 = (const float*)d_in[15];

    // Workspace layout (all offsets 64B-aligned), ~84.5 MB total.
    char* ws = (char*)d_ws;
    size_t off = 0;
    _Float16* xh  = (_Float16*)(ws + off); off += (size_t)kCin  * kN    * sizeof(_Float16); // 48 MB
    _Float16* y1h = (_Float16*)(ws + off); off += (size_t)kCmid * kN    * sizeof(_Float16); // 32 MB
    _Float16* W1p = (_Float16*)(ws + off); off += (size_t)kCmid * kCin  * sizeof(_Float16);
    _Float16* W2p = (_Float16*)(ws + off); off += (size_t)kCmid * kCmid * sizeof(_Float16);
    float* scale1 = (float*)(ws + off); off += kCmid * sizeof(float);
    float* bias1  = (float*)(ws + off); off += kCmid * sizeof(float);
    float* scale2 = (float*)(ws + off); off += kCmid * sizeof(float);
    float* bias2  = (float*)(ws + off); off += kCmid * sizeof(float);

    // BN folding + weight packing (tiny).
    fold_bn_kernel<<<1, 256, 0, stream>>>(b1, g1, be1, rm1, rv1, scale1, bias1, kCmid);
    fold_bn_kernel<<<1, 256, 0, stream>>>(b2, g2, be2, rm2, rv2, scale2, bias2, kCmid);
    {
        int t1 = (kCmid >> 4) * (kCin  >> 5) * 32;  // 6144
        int t2 = (kCmid >> 4) * (kCmid >> 5) * 32;  // 4096
        pack_w_kernel<<<(t1 + 255) / 256, 256, 0, stream>>>(W1, W1p, kCmid, kCin);
        pack_w_kernel<<<(t2 + 255) / 256, 256, 0, stream>>>(W2, W2p, kCmid, kCmid);
    }

    // 3-NN interpolation + concat -> x (f16, [384, 65536]).
    knn_interp_kernel<<<dim3(kNL / 256, kB), 256, 0, stream>>>(
        pc_down, pc_up, feat_down, feat_up, xh);

    // Layer 1: [256x384] @ [384x65536] + BN + ReLU -> f16 y1.
    gemm_bn_relu_kernel<kCin, 0><<<kN / 64, 256, 0, stream>>>(
        W1p, xh, scale1, bias1, y1h, nullptr);

    // Layer 2: [256x256] @ [256x65536] + BN + ReLU -> f32 out [B,256,NL].
    gemm_bn_relu_kernel<kCmid, 1><<<kN / 64, 256, 0, stream>>>(
        W2p, y1h, scale2, bias2, nullptr, (float*)d_out);
}
